// Attention_71751723647721
// MI455X (gfx1250) — matrix operand order repaired and checked by reading, not compiled, not run on hardware
//
#include <hip/hip_runtime.h>
#include <hip/hip_bf16.h>

typedef __attribute__((ext_vector_type(16))) _Float16 v16h;
typedef __attribute__((ext_vector_type(8)))  float    v8f;

#define HEADS_C   16
#define DH_C      64
#define DIM_C     1024
#define B_C       8
#define N_C       512
#define HB_C      128           // HEADS*B
#define ROWS1_C   4096          // B*N
#define TOTROWS_C 12288         // 3*B*N
#define NTOT_D    33554432.0    // HB*N*N elements in score tensors
#define INV_SQRT64 (1.0f/(8.0f+1e-6f))

// ---------------------------------------------------------------------------
// Register-blocked WMMA K-loop: one A fragment vs four B fragments (16x64 per
// wave).  4 v_wmma_f32_16x16x32_f16 per 5 fragment loads + global_prefetch_b8.
// ---------------------------------------------------------------------------
__device__ __forceinline__ void wmma_kloop4(const v16h* __restrict__ A,
                                            const v16h* __restrict__ B,
                                            int Ktiles, v8f acc[4]) {
  const size_t ns = (size_t)Ktiles * 32;   // v16h stride between n-tiles
  for (int kt = 0; kt < Ktiles; ++kt) {
    v16h a  = A[(size_t)kt * 32];
    v16h b0 = B[(size_t)kt * 32];
    v16h b1 = B[(size_t)kt * 32 + ns];
    v16h b2 = B[(size_t)kt * 32 + 2 * ns];
    v16h b3 = B[(size_t)kt * 32 + 3 * ns];
    if (kt + 1 < Ktiles) {
      __builtin_prefetch((const void*)(A + (size_t)(kt + 1) * 32), 0, 1);
      __builtin_prefetch((const void*)(B + (size_t)(kt + 1) * 32), 0, 1);
    }
    acc[0] = __builtin_amdgcn_wmma_f32_16x16x32_f16(false, a, false, b0, (short)0, acc[0], false, false);
    acc[1] = __builtin_amdgcn_wmma_f32_16x16x32_f16(false, a, false, b1, (short)0, acc[1], false, false);
    acc[2] = __builtin_amdgcn_wmma_f32_16x16x32_f16(false, a, false, b2, (short)0, acc[2], false, false);
    acc[3] = __builtin_amdgcn_wmma_f32_16x16x32_f16(false, a, false, b3, (short)0, acc[3], false, false);
  }
}

// packed-A index for element (row R, col k) of a matrix with Ktiles k-tiles
__device__ __forceinline__ size_t packA_idx(int R, int k, int Ktiles) {
  int mt = R >> 4, kt = k >> 5, kl = k & 31;
  int lane = (((kl >> 3) & 1) << 4) | (R & 15);
  int j = ((kl >> 4) << 2) | ((kl & 7) >> 1);
  return (((size_t)mt * Ktiles + kt) * 32 + lane) * 16 + (size_t)(j * 2 + (kl & 1));
}
// packed-B index for element (k, col n)
__device__ __forceinline__ size_t packB_idx(int k, int n, int Ktiles) {
  int nt = n >> 4, kt = k >> 5, kl = k & 31;
  int lane = (((kl >> 3) & 1) << 4) | (n & 15);
  int j = ((kl >> 4) << 2) | ((kl & 7) >> 1);
  return (((size_t)nt * Ktiles + kt) * 32 + lane) * 16 + (size_t)(j * 2 + (kl & 1));
}

// ---------------------------------------------------------------------------
// Generic batched WMMA GEMM (16x64 per wave), f32 row-major C
// ---------------------------------------------------------------------------
__global__ void gemm_f16(const _Float16* __restrict__ Apk, const _Float16* __restrict__ Bpk,
                         float* __restrict__ C, int Mtiles, int Ntiles4, int Ktiles, int ldc,
                         size_t strideA, size_t strideB, size_t strideC) {
  int lane = threadIdx.x & 31;
  int gw = (int)((blockIdx.x * blockDim.x + threadIdx.x) >> 5);
  if (gw >= Mtiles * Ntiles4) return;
  int mt = gw / Ntiles4, nt4 = gw - mt * Ntiles4;
  int bz = blockIdx.y;
  const v16h* A = (const v16h*)(Apk + (size_t)bz * strideA) + (size_t)mt * Ktiles * 32 + lane;
  const v16h* B = (const v16h*)(Bpk + (size_t)bz * strideB) + (size_t)(nt4 * 4) * Ktiles * 32 + lane;
  v8f z = {};
  v8f acc[4] = {z, z, z, z};
  wmma_kloop4(A, B, Ktiles, acc);
  float* Cb = C + (size_t)bz * strideC;
  int r0 = mt * 16 + ((lane >> 4) << 3);
#pragma unroll
  for (int j = 0; j < 4; ++j) {
    int col = (nt4 * 4 + j) * 16 + (lane & 15);
#pragma unroll
    for (int i = 0; i < 8; ++i) Cb[(size_t)(r0 + i) * ldc + col] = acc[j][i];
  }
}

// Final GEMM with bias epilogue (out @ w_out + b_out)
__global__ void gemm_f16_bias(const _Float16* __restrict__ Apk, const _Float16* __restrict__ Bpk,
                              const float* __restrict__ bias, float* __restrict__ C,
                              int Mtiles, int Ntiles4, int Ktiles, int ldc) {
  int lane = threadIdx.x & 31;
  int gw = (int)((blockIdx.x * blockDim.x + threadIdx.x) >> 5);
  if (gw >= Mtiles * Ntiles4) return;
  int mt = gw / Ntiles4, nt4 = gw - mt * Ntiles4;
  const v16h* A = (const v16h*)Apk + (size_t)mt * Ktiles * 32 + lane;
  const v16h* B = (const v16h*)Bpk + (size_t)(nt4 * 4) * Ktiles * 32 + lane;
  v8f z = {};
  v8f acc[4] = {z, z, z, z};
  wmma_kloop4(A, B, Ktiles, acc);
  int r0 = mt * 16 + ((lane >> 4) << 3);
#pragma unroll
  for (int j = 0; j < 4; ++j) {
    int col = (nt4 * 4 + j) * 16 + (lane & 15);
    float bs = bias[col];
#pragma unroll
    for (int i = 0; i < 8; ++i) C[(size_t)(r0 + i) * ldc + col] = acc[j][i] + bs;
  }
}

// attn @ f_v : one wave covers the full head dim (N=64 = 4 tiles); epilogue
// writes f16 directly into packed-A layout of the [4096 x 1024] output matrix
__global__ void gemm_av_pack(const _Float16* __restrict__ Apk, const _Float16* __restrict__ Bpk,
                             _Float16* __restrict__ OutPk) {
  const int Mtiles = 32, Ktiles = 16;
  int lane = threadIdx.x & 31;
  int mt = (int)((blockIdx.x * blockDim.x + threadIdx.x) >> 5);
  if (mt >= Mtiles) return;
  int hb = blockIdx.y, h = hb >> 3, b = hb & 7;
  const v16h* A = (const v16h*)(Apk + (size_t)hb * 262144) + (size_t)mt * Ktiles * 32 + lane;
  const v16h* B = (const v16h*)(Bpk + (size_t)hb * 32768) + lane;
  v8f z = {};
  v8f acc[4] = {z, z, z, z};
  wmma_kloop4(A, B, Ktiles, acc);
  int n0 = mt * 16 + ((lane >> 4) << 3);
#pragma unroll
  for (int j = 0; j < 4; ++j) {
    int d = j * 16 + (lane & 15);
    int colG = h * 64 + d;
#pragma unroll
    for (int i = 0; i < 8; ++i) {
      int rowG = b * 512 + n0 + i;
      OutPk[packA_idx(rowG, colG, 32)] = (_Float16)acc[j][i];
    }
  }
}

// ---------------------------------------------------------------------------
// Packing / preprocessing kernels
// ---------------------------------------------------------------------------
__global__ void pack_weightB(const float* __restrict__ W, _Float16* __restrict__ Bpk,
                             int K, int Nn) {
  int idx = blockIdx.x * blockDim.x + threadIdx.x;
  if (idx >= K * Nn) return;
  int k = idx / Nn, n = idx - k * Nn;
  Bpk[packB_idx(k, n, K >> 5)] = (_Float16)W[idx];
}

// LayerNorm over DIM=1024 then pack fp16 into packed-A (row offset selects q/k/v block)
__global__ void ln_pack(const float* __restrict__ X, const float* __restrict__ g,
                        const float* __restrict__ bb, _Float16* __restrict__ Apk, int rowOffset) {
  int r = blockIdx.x, t = threadIdx.x;
  __shared__ float red[256];
  float x[4]; float s = 0.f;
#pragma unroll
  for (int i = 0; i < 4; ++i) { x[i] = X[(size_t)r * 1024 + t * 4 + i]; s += x[i]; }
  red[t] = s; __syncthreads();
  for (int o = 128; o > 0; o >>= 1) { if (t < o) red[t] += red[t + o]; __syncthreads(); }
  float mean = red[0] * (1.f / 1024.f); __syncthreads();
  float vs = 0.f;
#pragma unroll
  for (int i = 0; i < 4; ++i) { float d = x[i] - mean; vs += d * d; }
  red[t] = vs; __syncthreads();
  for (int o = 128; o > 0; o >>= 1) { if (t < o) red[t] += red[t + o]; __syncthreads(); }
  float inv = rsqrtf(red[0] * (1.f / 1024.f) + 1e-5f);
  int R = rowOffset + r;
#pragma unroll
  for (int i = 0; i < 4; ++i) {
    int k = t * 4 + i;
    float y = (x[i] - mean) * inv * g[k] + bb[k];
    Apk[packA_idx(R, k, 32)] = (_Float16)y;
  }
}

// pack f_q rows into per-(h,b) A fragments (M=n 512, K=d 64, Ktiles=2)
__global__ void pack_qA(const float* __restrict__ F, _Float16* __restrict__ Apk) {
  int idx = blockIdx.x * blockDim.x + threadIdx.x;
  if (idx >= HB_C * N_C * DH_C) return;
  int hb = idx >> 15, rem = idx & 32767;
  int n = rem >> 6, d = rem & 63;
  int h = hb >> 3, b = hb & 7;
  float v = F[(size_t)(b * 512 + n) * 1024 + h * 64 + d];
  Apk[(size_t)hb * 32768 + packA_idx(n, d, 2)] = (_Float16)v;
}
// pack f_k into B fragments (K=d 64, N=m 512)
__global__ void pack_kB(const float* __restrict__ F, _Float16* __restrict__ Bpk) {
  int idx = blockIdx.x * blockDim.x + threadIdx.x;
  if (idx >= HB_C * N_C * DH_C) return;
  int hb = idx >> 15, rem = idx & 32767;
  int m = rem >> 6, d = rem & 63;
  int h = hb >> 3, b = hb & 7;
  float v = F[(size_t)(4096 + b * 512 + m) * 1024 + h * 64 + d];
  Bpk[(size_t)hb * 32768 + packB_idx(d, m, 2)] = (_Float16)v;
}
// pack f_v into B fragments (K=m 512, N=d 64)
__global__ void pack_vB(const float* __restrict__ F, _Float16* __restrict__ Bpk) {
  int idx = blockIdx.x * blockDim.x + threadIdx.x;
  if (idx >= HB_C * N_C * DH_C) return;
  int hb = idx >> 15, rem = idx & 32767;
  int m = rem >> 6, d = rem & 63;
  int h = hb >> 3, b = hb & 7;
  float v = F[(size_t)(8192 + b * 512 + m) * 1024 + h * 64 + d];
  Bpk[(size_t)hb * 32768 + packB_idx(m, d, 16)] = (_Float16)v;
}

// ---------------------------------------------------------------------------
// Reduction / scalar kernels
// ---------------------------------------------------------------------------
__global__ void k_colmean(const float* __restrict__ F, float* __restrict__ kbar) {
  int idx = blockIdx.x * blockDim.x + threadIdx.x;
  if (idx >= HB_C * DH_C) return;
  int h = idx >> 9, b = (idx >> 6) & 7, d = idx & 63;
  const float* p = F + (size_t)(4096 + b * 512) * 1024 + h * 64 + d;
  float s = 0.f;
  for (int m = 0; m < 512; ++m) s += p[(size_t)m * 1024];
  kbar[idx] = s * (1.f / 512.f);
}
__global__ void k_Ssum(const float* __restrict__ kbar, float* __restrict__ Sarr) {
  int hb = blockIdx.x * blockDim.x + threadIdx.x;
  if (hb >= HB_C) return;
  float s = 0.f;
  for (int d = 0; d < 64; ++d) s += kbar[hb * 64 + d];
  Sarr[hb] = s;
}
__global__ void k_krow(const float* __restrict__ F, float* __restrict__ nrmk,
                       float* __restrict__ skA) {
  int idx = blockIdx.x * blockDim.x + threadIdx.x;
  if (idx >= HB_C * N_C) return;
  int hb = idx >> 9, m = idx & 511, h = hb >> 3, b = hb & 7;
  const float* p = F + (size_t)(4096 + b * 512 + m) * 1024 + h * 64;
  float s = 0.f, q = 0.f;
  for (int d = 0; d < 64; ++d) { float x = p[d]; s += x; q += x * x; }
  nrmk[idx] = sqrtf(q); skA[idx] = s;
}
__global__ void k_qrow(const float* __restrict__ F, const float* __restrict__ kbar,
                       float* __restrict__ nrmq, float* __restrict__ muq,
                       float* __restrict__ tA) {
  int idx = blockIdx.x * blockDim.x + threadIdx.x;
  if (idx >= HB_C * N_C) return;
  int hb = idx >> 9, n = idx & 511, h = hb >> 3, b = hb & 7;
  const float* p  = F + (size_t)(b * 512 + n) * 1024 + h * 64;
  const float* kb = kbar + hb * 64;
  float s = 0.f, q = 0.f, t = 0.f;
  for (int d = 0; d < 64; ++d) { float x = p[d]; s += x; q += x * x; t += x * kb[d]; }
  nrmq[idx] = sqrtf(q); muq[idx] = s * (1.f / 64.f); tA[idx] = t;
}
__global__ void k_headmean(const float* __restrict__ F, float* __restrict__ hm) {
  int c = blockIdx.x * blockDim.x + threadIdx.x;
  if (c >= 2048) return;
  int col = c & 1023; size_t rowbase = (c < 1024) ? 0 : 4096;
  float s = 0.f;
  for (int r = 0; r < 4096; ++r) s += F[(rowbase + r) * 1024 + col];
  hm[c] = s * (1.f / 4096.f);
}

__device__ __forceinline__ float block_sum(float v, float* sh) {
  int t = threadIdx.x;
  sh[t] = v; __syncthreads();
  for (int o = 128; o > 0; o >>= 1) { if (t < o) sh[t] += sh[t + o]; __syncthreads(); }
  float r = sh[0]; __syncthreads();
  return r;
}

// pass 1 over dots: global cos/cov sums (f64 atomics) + per-row margin mean
__global__ void k_stats(const float* __restrict__ dots, const float* __restrict__ nrmq,
                        const float* __restrict__ muqA, const float* __restrict__ tA,
                        const float* __restrict__ nrmk, const float* __restrict__ skA,
                        const float* __restrict__ Sarr, double* __restrict__ gs,
                        float* __restrict__ rowMargin) {
  int row = blockIdx.x, t = threadIdx.x;
  int hb = row >> 9, n = row & 511;
  size_t base = (size_t)hb * 262144 + (size_t)n * 512;
  float nq = nrmq[row], mq = muqA[row], tn = tA[row], S = Sarr[hb];
  float cS = 0.f, cQ = 0.f, vS = 0.f, vQ = 0.f, mS = 0.f;
  for (int m = t; m < 512; m += 256) {
    float dt = dots[base + m];
    float nk = nrmk[hb * 512 + m], skm = skA[hb * 512 + m];
    float cosv = fminf(fmaxf(dt / ((nq + 1e-8f) * (nk + 1e-8f)), -0.98f), 0.98f);
    float csv  = fminf(fmaxf(dt / (fmaxf(nq, 1e-6f) * fmaxf(nk, 1e-6f)), -0.98f), 0.98f);
    float cov  = (dt - mq * skm - tn + mq * S) * INV_SQRT64;
    cS += cosv; cQ += cosv * cosv; vS += cov; vQ += cov * cov;
    mS += fminf(fmaxf(0.01f - csv, 0.f), 5.f);
  }
  __shared__ float sh[256];
  float a0 = block_sum(cS, sh), a1 = block_sum(cQ, sh);
  float a2 = block_sum(vS, sh), a3 = block_sum(vQ, sh);
  float a4 = block_sum(mS, sh);
  if (t == 0) {
    atomicAdd(&gs[0], (double)a0); atomicAdd(&gs[1], (double)a1);
    atomicAdd(&gs[2], (double)a2); atomicAdd(&gs[3], (double)a3);
    rowMargin[row] = a4 * (1.f / 512.f);
  }
}

// compute normalizers cn/covn/varn (+reg) from global sums
__global__ void k_norms(const double* __restrict__ gs, const float* __restrict__ rowMargin,
                        float* __restrict__ Gf) {
  __shared__ float s1[256], s2[256];
  int t = threadIdx.x;
  float s = 0.f, q = 0.f;
  for (int i = t; i < HB_C * N_C; i += 256) { float v = rowMargin[i]; s += v; q += v * v; }
  s1[t] = s; s2[t] = q; __syncthreads();
  for (int o = 128; o > 0; o >>= 1) { if (t < o) { s1[t] += s1[t + o]; s2[t] += s2[t + o]; } __syncthreads(); }
  if (t == 0) {
    double Ntot = NTOT_D;
    double stdc = sqrt(fmax(0.0, (gs[1] - gs[0] * gs[0] / Ntot) / (Ntot - 1.0)));
    double health = sqrt(fmax(0.0, (gs[3] - gs[2] * gs[2] / Ntot) / (Ntot - 1.0)));
    float basef = 0.001f / 512.f;
    float reg = (health < 1e-5) ? basef * 5.f : ((health < 1e-3) ? basef * 2.f : basef);
    float cn = (float)stdc + 1e-6f;
    float covn = reg * (float)health + 1e-6f;   // clip ±25 cannot bind at reg~2e-6
    double ms = 512.0 * (double)s1[0], mqq = 512.0 * (double)s2[0];
    double stdv = sqrt(fmax(0.0, (mqq - ms * ms / Ntot) / (Ntot - 1.0)));
    float varn = (float)stdv + 1e-6f;
    Gf[0] = cn;   Gf[1] = fminf(cn, 1.f);
    Gf[2] = covn; Gf[3] = fminf(covn * 10.f, 1.f);
    Gf[4] = varn; Gf[5] = fminf(varn * 10.f, 1.f);
    Gf[6] = reg;
  }
}

// per-head weight-predictor MLP (tiny; one lane per head)
__global__ void k_mlp(const float* __restrict__ hm, const float* __restrict__ w1,
                      const float* __restrict__ b1, const float* __restrict__ lng,
                      const float* __restrict__ lnb, const float* __restrict__ w2,
                      const float* __restrict__ b2, const float* __restrict__ w3,
                      const float* __restrict__ b3, const float* __restrict__ wtemp,
                      float* __restrict__ Wts) {
  int h = threadIdx.x;
  if (h >= 16) return;
  float qk[128];
  for (int i = 0; i < 64; ++i) { qk[i] = hm[h * 64 + i]; qk[64 + i] = hm[1024 + h * 64 + i]; }
  float h1[64];
  for (int j = 0; j < 64; ++j) {
    float a = b1[j];
    for (int i = 0; i < 128; ++i) a += qk[i] * w1[i * 64 + j];
    h1[j] = a;
  }
  float mu = 0.f; for (int j = 0; j < 64; ++j) mu += h1[j]; mu *= (1.f / 64.f);
  float var = 0.f; for (int j = 0; j < 64; ++j) { float d = h1[j] - mu; var += d * d; }
  float inv = rsqrtf(var * (1.f / 64.f) + 1e-5f);
  for (int j = 0; j < 64; ++j) h1[j] = fmaxf((h1[j] - mu) * inv * lng[j] + lnb[j], 0.f);
  float h2[32];
  for (int j = 0; j < 32; ++j) {
    float a = b2[j];
    for (int i = 0; i < 64; ++i) a += h1[i] * w2[i * 32 + j];
    h2[j] = fmaxf(a, 0.f);
  }
  float lg[3];
  for (int j = 0; j < 3; ++j) {
    float a = b3[j];
    for (int i = 0; i < 32; ++i) a += h2[i] * w3[i * 3 + j];
    lg[j] = a;
  }
  float mx = fmaxf(lg[0], fmaxf(lg[1], lg[2]));
  float e[3]; float se = 0.f;
  for (int j = 0; j < 3; ++j) { e[j] = expf(lg[j] - mx); se += e[j]; }
  float lo[3]; for (int j = 0; j < 3; ++j) lo[j] = e[j] / se;   // first softmax
  float temp = fminf(fmaxf(wtemp[0], 0.1f), 5.f);
  float y0 = lo[0] / temp, y1 = lo[1] / temp, y2 = lo[2] / temp;
  float mx2 = fmaxf(y0, fmaxf(y1, y2));
  float f0 = expf(y0 - mx2), f1 = expf(y1 - mx2), f2 = expf(y2 - mx2);
  float sf = f0 + f1 + f2;
  float w0 = fminf(fmaxf(f0 / sf, 0.08f), 0.75f);
  float w1c = fminf(fmaxf(f1 / sf, 0.08f), 0.75f);
  float w2c = fminf(fmaxf(f2 / sf, 0.08f), 0.75f);
  float ws = w0 + w1c + w2c;
  Wts[h * 3 + 0] = w0 / ws; Wts[h * 3 + 1] = w1c / ws; Wts[h * 3 + 2] = w2c / ws;
}

// pass 2: dots -> dots_f in place; accumulate sums for std(dots_f)
__global__ void k_combine(float* __restrict__ dots, const float* __restrict__ nrmq,
                          const float* __restrict__ muqA, const float* __restrict__ tA,
                          const float* __restrict__ nrmk, const float* __restrict__ skA,
                          const float* __restrict__ Sarr, const float* __restrict__ rowMargin,
                          const float* __restrict__ Gf, const float* __restrict__ Wts,
                          double* __restrict__ gs) {
  int row = blockIdx.x, t = threadIdx.x;
  int hb = row >> 9, n = row & 511, h = hb >> 3;
  size_t base = (size_t)hb * 262144 + (size_t)n * 512;
  float nq = nrmq[row], mq = muqA[row], tn = tA[row], S = Sarr[hb];
  float cw = Wts[h * 3 + 0], covw = Wts[h * 3 + 1], varw = Wts[h * 3 + 2];
  float cn = Gf[0], ch = Gf[1], covn = Gf[2], covh = Gf[3];
  float varn = Gf[4], varh = Gf[5], reg = Gf[6];
  float varterm = varw * (rowMargin[row] / varn * 0.4f * varh);
  float dS = 0.f, dQ = 0.f;
  for (int m = t; m < 512; m += 256) {
    float dt = dots[base + m];
    float nk = nrmk[hb * 512 + m], skm = skA[hb * 512 + m];
    float cosv = fminf(fmaxf(dt / ((nq + 1e-8f) * (nk + 1e-8f)), -0.98f), 0.98f);
    float covp = (dt - mq * skm - tn + mq * S) * INV_SQRT64;
    float cov  = fminf(fmaxf(reg * covp, -25.f), 25.f);
    float df = cw * (cosv / cn * ch) + covw * (cov / covn * 0.4f * covh) + varterm;
    dots[base + m] = df;
    dS += df; dQ += df * df;
  }
  __shared__ float sh[256];
  float a0 = block_sum(dS, sh), a1 = block_sum(dQ, sh);
  if (t == 0) { atomicAdd(&gs[4], (double)a0); atomicAdd(&gs[5], (double)a1); }
}

__global__ void k_atemp(const double* __restrict__ gs, float* __restrict__ Gf) {
  if (threadIdx.x == 0) {
    double Ntot = NTOT_D;
    double div = sqrt(fmax(0.0, (gs[5] - gs[4] * gs[4] / Ntot) / (Ntot - 1.0)));
    float at = (div < 1e-5) ? 0.05f : ((div < 1e-3) ? 0.2f : 0.5f + (float)div);
    Gf[7] = fminf(fmaxf(at, 0.05f), 2.f);
  }
}

// softmax over keys, emit f16 probs directly in packed-A layout (M=n, K=m, Ktiles=16)
__global__ void k_softmax_pack(const float* __restrict__ dots, const float* __restrict__ Gf,
                               _Float16* __restrict__ Apk) {
  int row = blockIdx.x, t = threadIdx.x;
  int hb = row >> 9, n = row & 511;
  size_t base = (size_t)hb * 262144 + (size_t)n * 512;
  float inv_at = 1.f / Gf[7];
  float x0 = dots[base + t] * inv_at;
  float x1 = dots[base + 256 + t] * inv_at;
  __shared__ float sh[256];
  sh[t] = fmaxf(x0, x1); __syncthreads();
  for (int o = 128; o > 0; o >>= 1) { if (t < o) sh[t] = fmaxf(sh[t], sh[t + o]); __syncthreads(); }
  float mx = sh[0]; __syncthreads();
  float e0 = expf(x0 - mx), e1 = expf(x1 - mx);
  sh[t] = e0 + e1; __syncthreads();
  for (int o = 128; o > 0; o >>= 1) { if (t < o) sh[t] += sh[t + o]; __syncthreads(); }
  float inv = 1.f / sh[0];
  size_t slice = (size_t)hb * 262144;
  Apk[slice + packA_idx(n, t, 16)]       = (_Float16)(e0 * inv);
  Apk[slice + packA_idx(n, t + 256, 16)] = (_Float16)(e1 * inv);
}

// ---------------------------------------------------------------------------
extern "C" void kernel_launch(void* const* d_in, const int* in_sizes, int n_in,
                              void* d_out, int out_size, void* d_ws, size_t ws_size,
                              hipStream_t stream) {
  const float* q     = (const float*)d_in[0];
  const float* k     = (const float*)d_in[1];
  const float* v     = (const float*)d_in[2];
  const float* ln_g  = (const float*)d_in[3];
  const float* ln_b  = (const float*)d_in[4];
  const float* w_in  = (const float*)d_in[5];
  const float* wp_w1 = (const float*)d_in[6];
  const float* wp_b1 = (const float*)d_in[7];
  const float* wp_lng= (const float*)d_in[8];
  const float* wp_lnb= (const float*)d_in[9];
  const float* wp_w2 = (const float*)d_in[10];
  const float* wp_b2 = (const float*)d_in[11];
  const float* wp_w3 = (const float*)d_in[12];
  const float* wp_b3 = (const float*)d_in[13];
  const float* w_temp= (const float*)d_in[14];
  const float* w_out = (const float*)d_in[15];
  const float* b_out = (const float*)d_in[16];
  float* out = (float*)d_out;

  char* ws = (char*)d_ws; size_t off = 0;
  auto carve = [&](size_t bytes) -> char* {
    char* p = ws + off; off = (off + bytes + 255) & ~((size_t)255); return p;
  };
  _Float16* pA_in   = (_Float16*)carve((size_t)TOTROWS_C * DIM_C * 2);
  _Float16* pB_win  = (_Float16*)carve((size_t)DIM_C * DIM_C * 2);
  _Float16* pB_wout = (_Float16*)carve((size_t)DIM_C * DIM_C * 2);
  float*    F       = (float*)carve((size_t)TOTROWS_C * DIM_C * 4);
  float*    dots    = (float*)carve((size_t)HB_C * N_C * N_C * 4);
  _Float16* pA_q    = (_Float16*)carve((size_t)HB_C * 32768 * 2);
  _Float16* pB_k    = (_Float16*)carve((size_t)HB_C * 32768 * 2);
  _Float16* pB_v    = (_Float16*)carve((size_t)HB_C * 32768 * 2);
  _Float16* pA_attn = (_Float16*)carve((size_t)HB_C * 262144 * 2);
  _Float16* pA_out  = (_Float16*)carve((size_t)ROWS1_C * DIM_C * 2);
  float*    kbar    = (float*)carve(HB_C * DH_C * 4);
  float*    Sarr    = (float*)carve(HB_C * 4);
  float*    nrmq    = (float*)carve(HB_C * N_C * 4);
  float*    muq     = (float*)carve(HB_C * N_C * 4);
  float*    tA      = (float*)carve(HB_C * N_C * 4);
  float*    nrmk    = (float*)carve(HB_C * N_C * 4);
  float*    skA     = (float*)carve(HB_C * N_C * 4);
  float*    hm      = (float*)carve(2048 * 4);
  float*    rowMargin=(float*)carve(HB_C * N_C * 4);
  double*   gs      = (double*)carve(8 * 8);
  float*    Gf      = (float*)carve(16 * 4);
  float*    Wts     = (float*)carve(HEADS_C * 3 * 4);

  hipMemsetAsync(gs, 0, 8 * sizeof(double), stream);

  // weight packing
  pack_weightB<<<4096, 256, 0, stream>>>(w_in,  pB_win,  DIM_C, DIM_C);
  pack_weightB<<<4096, 256, 0, stream>>>(w_out, pB_wout, DIM_C, DIM_C);
  // LayerNorm + pack A for q,k,v
  ln_pack<<<ROWS1_C, 256, 0, stream>>>(q, ln_g, ln_b, pA_in, 0);
  ln_pack<<<ROWS1_C, 256, 0, stream>>>(k, ln_g, ln_b, pA_in, 4096);
  ln_pack<<<ROWS1_C, 256, 0, stream>>>(v, ln_g, ln_b, pA_in, 8192);
  // projection GEMM: [12288,1024] = LN(x) @ w_in  (Mtiles=768, Ntiles4=16, Ktiles=32)
  gemm_f16<<<dim3(1536, 1), 256, 0, stream>>>(pA_in, pB_win, F, 768, 16, 32, 1024, 0, 0, 0);
  // per-row/col/head reductions on F
  k_colmean<<<32, 256, 0, stream>>>(F, kbar);
  k_Ssum<<<1, 128, 0, stream>>>(kbar, Sarr);
  k_krow<<<256, 256, 0, stream>>>(F, nrmk, skA);
  k_qrow<<<256, 256, 0, stream>>>(F, kbar, nrmq, muq, tA);
  k_headmean<<<8, 256, 0, stream>>>(F, hm);
  // repack per-(h,b) fragments
  pack_qA<<<16384, 256, 0, stream>>>(F, pA_q);
  pack_kB<<<16384, 256, 0, stream>>>(F, pB_k);
  pack_vB<<<16384, 256, 0, stream>>>(F, pB_v);
  // dots = f_q @ f_k^T, batched over 128 (h,b): Mtiles=32, Ntiles4=8, Ktiles=2
  gemm_f16<<<dim3(32, HB_C), 256, 0, stream>>>(pA_q, pB_k, dots, 32, 8, 2, 512,
                                               32768, 32768, 262144);
  // global statistics pass
  k_stats<<<HB_C * N_C, 256, 0, stream>>>(dots, nrmq, muq, tA, nrmk, skA, Sarr, gs, rowMargin);
  k_norms<<<1, 256, 0, stream>>>(gs, rowMargin, Gf);
  k_mlp<<<1, 32, 0, stream>>>(hm, wp_w1, wp_b1, wp_lng, wp_lnb, wp_w2, wp_b2,
                              wp_w3, wp_b3, w_temp, Wts);
  // combine into dots_f (in place) + std(dots_f) sums
  k_combine<<<HB_C * N_C, 256, 0, stream>>>(dots, nrmq, muq, tA, nrmk, skA, Sarr,
                                            rowMargin, Gf, Wts, gs);
  k_atemp<<<1, 32, 0, stream>>>(gs, Gf);
  // softmax + pack attn probs
  k_softmax_pack<<<HB_C * N_C, 256, 0, stream>>>(dots, Gf, pA_attn);
  // attn @ f_v (one wave covers N=64), epilogue packs into final-GEMM A fragments
  gemm_av_pack<<<dim3(4, HB_C), 256, 0, stream>>>(pA_attn, pB_v, pA_out);
  // out @ w_out + b_out: Mtiles=256, Ntiles4=16, Ktiles=32 -> d_out [4096,1024] f32
  gemm_f16_bias<<<512, 256, 0, stream>>>(pA_out, pB_wout, b_out, out, 256, 16, 32, 1024);
}
